// Involution_75282186764558
// MI455X (gfx1250) — matrix operand order repaired and checked by reading, not compiled, not run on hardware
//
#include <hip/hip_runtime.h>
#include <stdint.h>

// ---------------- types matching gfx1250 WMMA builtins ----------------
typedef __attribute__((ext_vector_type(16))) __bf16 v16bf;
typedef __attribute__((ext_vector_type(8)))  float  v8f;

#define BB   16
#define CC   256
#define CRD  128
#define HH   64
#define WW_  64
#define HWP  (HH * WW_)        // 4096
#define NTOT (BB * CC * HWP)   // 16,777,216
#define EPSV 1e-5f

// ---------------- bf16 helpers (RNE) ----------------
__device__ __forceinline__ uint16_t f32_to_bf16(float f) {
  union { float f; uint32_t u; } v; v.f = f;
  uint32_t u = v.u;
  uint32_t r = (u + 0x7FFFu + ((u >> 16) & 1u)) >> 16;
  return (uint16_t)r;
}
__device__ __forceinline__ float bf16_to_f32(uint16_t h) {
  union { uint32_t u; float f; } v; v.u = ((uint32_t)h) << 16;
  return v.f;
}

// ---------------- WMMA fragment loaders --------------------------------
// A-operand (16x32 bf16, row-major source): lane ln = row M; VGPR0-7 hold
// K = half*16..+15 (contiguous, matches B)  -> two 16B loads.
__device__ __forceinline__ v16bf ld_a(const uint16_t* row, int k0, int half) {
  union { uint4 u[2]; v16bf v; } t;
  t.u[0] = *(const uint4*)(row + k0 + half * 16);
  t.u[1] = *(const uint4*)(row + k0 + half * 16 + 8);
  return t.v;
}
// B-operand (32x16 bf16, pixel-major source): lane ln = col N; VGPR r holds
// K = half*16 + 2r,2r+1 -> 16 consecutive bf16 = 32 contiguous bytes.
__device__ __forceinline__ v16bf ld_b(const uint16_t* p) {
  union { uint4 u[2]; v16bf v; } t;
  t.u[0] = *(const uint4*)(p);
  t.u[1] = *(const uint4*)(p + 8);
  return t.v;
}
__device__ __forceinline__ v8f wmma_bf16(v16bf a, v16bf b, v8f c) {
  return __builtin_amdgcn_wmma_f32_16x16x32_bf16(false, a, false, b, (short)0, c,
                                                 false, false);
}

// ---------------- weight convert (zero-pad rows) ------------------------
__global__ void k_cvt_w(const float* __restrict__ src, uint16_t* __restrict__ dst,
                        int nsrc, int ntot) {
  int i = blockIdx.x * blockDim.x + threadIdx.x;
  if (i < ntot) dst[i] = (i < nsrc) ? f32_to_bf16(src[i]) : (uint16_t)0;
}

// ---------------- fused BN(+ReLU)+bf16 with LDS-tiled transpose ---------
// src: f32 channel-major [B][CC][HWP].
// cm (optional): bf16 channel-major.  pm: bf16 pixel-major [B][HWP][CC].
// 32x32 tile through LDS: global reads coalesced over pixels, pm writes
// 64B-contiguous over channels (no 2-byte scatter).
__global__ void k_bn_transpose(const float* __restrict__ src,
                               const float* __restrict__ scale,
                               const float* __restrict__ shift, int do_relu,
                               uint16_t* __restrict__ cm, uint16_t* __restrict__ pm) {
  __shared__ uint16_t tile[32][33];
  const int b  = blockIdx.z;
  const int c0 = blockIdx.y * 32;
  const int p0 = blockIdx.x * 32;
  const int tx = threadIdx.x;  // 0..31
  const int ty = threadIdx.y;  // 0..7
#pragma unroll
  for (int cy = ty; cy < 32; cy += 8) {
    int c = c0 + cy;
    size_t idx = ((size_t)b * CC + c) * HWP + p0 + tx;
    float v = src[idx];
    if (scale) v = v * scale[c] + shift[c];
    if (do_relu) v = v > 0.f ? v : 0.f;
    uint16_t hv = f32_to_bf16(v);
    if (cm) cm[idx] = hv;
    tile[cy][tx] = hv;
  }
  __syncthreads();
#pragma unroll
  for (int py = ty; py < 32; py += 8) {
    pm[((size_t)b * HWP + (p0 + py)) * CC + c0 + tx] = tile[tx][py];
  }
}

// ---------------- WMMA GEMM: 32(out) x 64(pixel) per wave ---------------
// out[o,p] = sum_k A[o,k] * Bpm[b][p][k] + bias[o]
// KD is compile-time: the K loop fully unrolls, every fragment address is
// base + constant (folded into the load's 24-bit ioffset), and the scheduler
// freely hoists global_load_b128 across earlier WMMAs -- no register-rotation
// movs, no per-iteration 64-bit address adds.
template <bool BF16PM, int KD>
__global__ void k_gemm(const uint16_t* __restrict__ A,
                       const uint16_t* __restrict__ Bpm,
                       const float* __restrict__ bias,
                       float* __restrict__ outCM,
                       uint16_t* __restrict__ outPM,
                       int Mreal) {
  const int b    = blockIdx.z;
  const int mo   = blockIdx.y * 32;
  const int np   = blockIdx.x * 64;
  const int lane = threadIdx.x;
  const int half = lane >> 4;
  const int ln   = lane & 15;

  const uint16_t* Arow0 = A + (size_t)(mo + ln) * KD;
  const uint16_t* Arow1 = A + (size_t)(mo + 16 + ln) * KD;
  const uint16_t* Brow[4];
#pragma unroll
  for (int t = 0; t < 4; ++t)
    Brow[t] = Bpm + ((size_t)b * HWP + (np + t * 16 + ln)) * KD + half * 16;

  v8f acc[2][4];
#pragma unroll
  for (int g = 0; g < 2; ++g)
#pragma unroll
    for (int t = 0; t < 4; ++t)
#pragma unroll
      for (int e = 0; e < 8; ++e) acc[g][t][e] = 0.0f;

#pragma unroll
  for (int k0 = 0; k0 < KD; k0 += 32) {
    v16bf a0 = ld_a(Arow0, k0, half);
    v16bf a1 = ld_a(Arow1, k0, half);
#pragma unroll
    for (int t = 0; t < 4; ++t) {
      v16bf bt = ld_b(Brow[t] + k0);
      acc[0][t] = wmma_bf16(a0, bt, acc[0][t]);
      acc[1][t] = wmma_bf16(a1, bt, acc[1][t]);
    }
  }

  if (BF16PM) {
    // full-tile store, pixel-major bf16 [B][HWP][Mreal]: 16B per store
#pragma unroll
    for (int g = 0; g < 2; ++g) {
#pragma unroll
      for (int t = 0; t < 4; ++t) {
        int p = np + t * 16 + ln;
        uint16_t pk[8];
#pragma unroll
        for (int r = 0; r < 8; ++r) {
          int o = mo + g * 16 + half * 8 + r;
          pk[r] = f32_to_bf16(acc[g][t][r] + bias[o]);
        }
        *(uint4*)(outPM + ((size_t)b * HWP + p) * Mreal + mo + g * 16 + half * 8) =
            *(const uint4*)pk;
      }
    }
  } else {
    // channel-major f32 [B][Mreal][HWP], guarded for Mreal < tile (span)
#pragma unroll
    for (int g = 0; g < 2; ++g) {
#pragma unroll
      for (int t = 0; t < 4; ++t) {
        int p = np + t * 16 + ln;
#pragma unroll
        for (int r = 0; r < 8; ++r) {
          int o = mo + g * 16 + half * 8 + r;
          if (o < Mreal)
            outCM[((size_t)b * Mreal + o) * HWP + p] = acc[g][t][r] + bias[o];
        }
      }
    }
  }
}

// ---------------- per-channel sum / sumsq reduction ---------------------
__global__ void k_stats(const float* __restrict__ src, float* __restrict__ sums,
                        float* __restrict__ sqs, int Cn) {
  int c = blockIdx.x;
  float s = 0.f, q = 0.f;
  for (int bb2 = 0; bb2 < BB; ++bb2) {
    const float* rowp = src + ((size_t)bb2 * Cn + c) * HWP;
    for (int p = threadIdx.x; p < HWP; p += 256) {
      float v = rowp[p];
      s += v; q += v * v;
    }
  }
  __shared__ float ss[256], sq[256];
  ss[threadIdx.x] = s; sq[threadIdx.x] = q;
  __syncthreads();
  for (int st = 128; st > 0; st >>= 1) {
    if ((int)threadIdx.x < st) {
      ss[threadIdx.x] += ss[threadIdx.x + st];
      sq[threadIdx.x] += sq[threadIdx.x + st];
    }
    __syncthreads();
  }
  if (threadIdx.x == 0) { sums[c] = ss[0]; sqs[c] = sq[0]; }
}

// ---------------- BN scale/shift from stats -----------------------------
__global__ void k_bnparams(const float* __restrict__ sums, const float* __restrict__ sqs,
                           const float* __restrict__ gamma, const float* __restrict__ beta,
                           float* __restrict__ scale, float* __restrict__ shift) {
  int c = threadIdx.x;
  float n   = (float)(BB * HWP);
  float mu  = sums[c] / n;
  float var = sqs[c] / n - mu * mu;
  float sc  = gamma[c] * rsqrtf(var + EPSV);
  scale[c] = sc;
  shift[c] = beta[c] - mu * sc;
}

// ---------------- involution: 9-tap dynamic-kernel gather ----------------
__global__ void k_invol(const uint16_t* __restrict__ hcm, const float* __restrict__ kern,
                        float* __restrict__ mid) {
  int i = blockIdx.x * blockDim.x + threadIdx.x;
  if (i >= NTOT) return;
  int p = i % HWP;
  int t = i / HWP;
  int c = t % CC;
  int b = t / CC;
  int y = p >> 6, x = p & 63;
  const float*    kb = kern + (size_t)b * 9 * HWP;
  const uint16_t* hb = hcm + ((size_t)b * CC + c) * HWP;
  float s = 0.f;
#pragma unroll
  for (int ky = 0; ky < 3; ++ky) {
    int yy = y + ky - 1;
    if (yy < 0 || yy >= HH) continue;
#pragma unroll
    for (int kx = 0; kx < 3; ++kx) {
      int xx = x + kx - 1;
      if (xx < 0 || xx >= WW_) continue;
      s += kb[(size_t)(ky * 3 + kx) * HWP + p] * bf16_to_f32(hb[yy * WW_ + xx]);
    }
  }
  mid[i] = s;
}

// ---------------- final: BN(post) + residual ----------------------------
__global__ void k_final(const float* __restrict__ post,
                        const float* __restrict__ scale, const float* __restrict__ shift,
                        const float* __restrict__ x, float* __restrict__ out) {
  int i = blockIdx.x * blockDim.x + threadIdx.x;
  if (i >= NTOT) return;
  int c = (i / HWP) % CC;
  out[i] = post[i] * scale[c] + shift[c] + x[i];
}

// ========================================================================
extern "C" void kernel_launch(void* const* d_in, const int* in_sizes, int n_in,
                              void* d_out, int out_size, void* d_ws, size_t ws_size,
                              hipStream_t stream) {
  const float* x          = (const float*)d_in[0];
  const float* pre_w      = (const float*)d_in[1];
  const float* pre_b      = (const float*)d_in[2];
  const float* pre_gamma  = (const float*)d_in[3];
  const float* pre_beta   = (const float*)d_in[4];
  const float* red_w      = (const float*)d_in[5];
  const float* red_b      = (const float*)d_in[6];
  const float* span_w     = (const float*)d_in[7];
  const float* span_b     = (const float*)d_in[8];
  const float* mid_gamma  = (const float*)d_in[9];
  const float* mid_beta   = (const float*)d_in[10];
  const float* post_w     = (const float*)d_in[11];
  const float* post_b     = (const float*)d_in[12];
  const float* post_gamma = (const float*)d_in[13];
  const float* post_beta  = (const float*)d_in[14];

  // -------- workspace carve-up (L2-resident working set, ~187 MB) ------
  char* ws = (char*)d_ws;
  size_t off = 0;
  auto take = [&](size_t bytes) -> char* {
    char* p = ws + off;
    off += (bytes + 255) & ~(size_t)255;
    return p;
  };
  uint16_t* xpm   = (uint16_t*)take((size_t)BB * HWP * CC * 2);   // 33.5 MB
  uint16_t* wpre  = (uint16_t*)take(256 * 256 * 2);
  uint16_t* wred  = (uint16_t*)take(128 * 256 * 2);
  uint16_t* wspan = (uint16_t*)take(32 * 128 * 2);                // 9 rows + zero pad to 32
  uint16_t* wpost = (uint16_t*)take(256 * 256 * 2);
  float*    fbuf  = (float*)take((size_t)NTOT * 4);               // 67 MB (pre->mid->post)
  uint16_t* hcm   = (uint16_t*)take((size_t)NTOT * 2);            // 33.5 MB
  uint16_t* hpm   = (uint16_t*)take((size_t)NTOT * 2);            // 33.5 MB (h_pm, then midh_pm)
  uint16_t* redpm = (uint16_t*)take((size_t)BB * HWP * CRD * 2);  // 16.8 MB
  float*    kern  = (float*)take((size_t)BB * 9 * HWP * 4);       // 2.4 MB
  float*    sums  = (float*)take(256 * 4);
  float*    sqs   = (float*)take(256 * 4);
  float*    scale = (float*)take(256 * 4);
  float*    shift = (float*)take(256 * 4);
  (void)ws_size; (void)in_sizes; (void)n_in; (void)out_size;

  const int  ELT_BLOCKS = (NTOT + 255) / 256;
  const dim3 TGRID(HWP / 32, CC / 32, BB);   // transpose tiles
  const dim3 TBLK(32, 8);

  // -------- 0. weight conversion + x transpose-to-pixel-major ----------
  k_cvt_w<<<(65536 + 255) / 256, 256, 0, stream>>>(pre_w,  wpre,  256 * 256, 256 * 256);
  k_cvt_w<<<(32768 + 255) / 256, 256, 0, stream>>>(red_w,  wred,  128 * 256, 128 * 256);
  k_cvt_w<<<(4096  + 255) / 256, 256, 0, stream>>>(span_w, wspan, 9 * 128,   32 * 128);
  k_cvt_w<<<(65536 + 255) / 256, 256, 0, stream>>>(post_w, wpost, 256 * 256, 256 * 256);
  k_bn_transpose<<<TGRID, TBLK, 0, stream>>>(x, nullptr, nullptr, 0, nullptr, xpm);

  // -------- 1. pre conv (256x256 GEMM, WMMA) -> fbuf (f32 CM) ----------
  k_gemm<false, 256><<<dim3(HWP / 64, 8, BB), 32, 0, stream>>>(
      wpre, xpm, pre_b, fbuf, nullptr, 256);

  // -------- 2. pre BN stats + params, BN+ReLU -> h (cm + pm bf16) ------
  k_stats<<<256, 256, 0, stream>>>(fbuf, sums, sqs, 256);
  k_bnparams<<<1, 256, 0, stream>>>(sums, sqs, pre_gamma, pre_beta, scale, shift);
  k_bn_transpose<<<TGRID, TBLK, 0, stream>>>(fbuf, scale, shift, 1, hcm, hpm);

  // -------- 3. reduce conv (128x256 GEMM) -> redpm (bf16 PM) -----------
  k_gemm<true, 256><<<dim3(HWP / 64, 4, BB), 32, 0, stream>>>(
      wred, hpm, red_b, nullptr, redpm, 128);

  // -------- 4. span conv (9x128 GEMM, M padded) -> kern (f32 CM) -------
  k_gemm<false, 128><<<dim3(HWP / 64, 1, BB), 32, 0, stream>>>(
      wspan, redpm, span_b, kern, nullptr, 9);

  // -------- 5. involution (dynamic 3x3 neighborhood sum) -> fbuf -------
  k_invol<<<ELT_BLOCKS, 256, 0, stream>>>(hcm, kern, fbuf);

  // -------- 6. mid BN stats + params, BN+ReLU -> midh (pm bf16) --------
  k_stats<<<256, 256, 0, stream>>>(fbuf, sums, sqs, 256);
  k_bnparams<<<1, 256, 0, stream>>>(sums, sqs, mid_gamma, mid_beta, scale, shift);
  k_bn_transpose<<<TGRID, TBLK, 0, stream>>>(fbuf, scale, shift, 1, nullptr, hpm);

  // -------- 7. post conv (256x256 GEMM) -> fbuf (f32 CM) ---------------
  k_gemm<false, 256><<<dim3(HWP / 64, 8, BB), 32, 0, stream>>>(
      wpost, hpm, post_b, fbuf, nullptr, 256);

  // -------- 8. post BN stats + params, BN + residual -> d_out ----------
  k_stats<<<256, 256, 0, stream>>>(fbuf, sums, sqs, 256);
  k_bnparams<<<1, 256, 0, stream>>>(sums, sqs, post_gamma, post_beta, scale, shift);
  k_final<<<ELT_BLOCKS, 256, 0, stream>>>(fbuf, scale, shift, x, (float*)d_out);
}
